// MeanAggregator_88742614270077
// MI455X (gfx1250) — matrix operand order repaired and checked by reading, not compiled, not run on hardware
//
#include <hip/hip_runtime.h>

typedef __attribute__((ext_vector_type(16))) _Float16 v16h;
typedef __attribute__((ext_vector_type(8)))  _Float16 v8h;
typedef __attribute__((ext_vector_type(8)))  float    v8f;

#define N_NODES 4096
#define M_NEIGH 8192
#define F_DIM   512
#define ALPHA   0.2f
#define PSTRIDE 40   // halves per P row in LDS: 80B => 20 dwords, conflict-free

// ---------------------------------------------------------------------------
// Helpers
// ---------------------------------------------------------------------------
__device__ __forceinline__ float att_p(float snr, float sm, int mk, float Bb) {
  float sc = snr + sm;
  sc = sc > 0.f ? sc : ALPHA * sc;          // LeakyReLU before mask (matches ref)
  float e = __expf(sc - Bb);                // Bb is an upper bound -> e <= 1
  return mk > 0 ? e : 0.f;                  // masked -> exp(NEG - Bb) == 0 exactly
}

// Producer step: one thread computes 2 adjacent probabilities of one score row
// and deposits them packed (half2) into the LDS P tile. Returns p0+p1.
__device__ __forceinline__ float p_block(const int* __restrict__ mrow,
                                         const float* __restrict__ s_m,
                                         int m0, int cpair,
                                         float snr, float Bb,
                                         _Float16* __restrict__ dst) {
  int2   mk = *(const int2*)  (mrow + m0 + cpair);
  float2 sm = *(const float2*)(s_m  + m0 + cpair);
  float p0 = att_p(snr, sm.x, mk.x, Bb);
  float p1 = att_p(snr, sm.y, mk.y, Bb);
  union { _Float16 h[2]; unsigned u; } pk;
  pk.h[0] = (_Float16)p0; pk.h[1] = (_Float16)p1;
  *(unsigned*)dst = pk.u;                    // single ds_store_b32
  return p0 + p1;
}

// ---------------------------------------------------------------------------
// Kernel C: bulk f32 -> f16 conversion (memory bound pre-pass)
// ---------------------------------------------------------------------------
__global__ void gat_cvt(const float* __restrict__ src,
                        _Float16* __restrict__ dst, int n) {
  int i = (blockIdx.x * blockDim.x + threadIdx.x) * 4;
  if (i >= n) return;
  float4 v = *(const float4*)(src + i);
  union { _Float16 h[4]; uint2 u; } pk;
  pk.h[0] = (_Float16)v.x; pk.h[1] = (_Float16)v.y;
  pk.h[2] = (_Float16)v.z; pk.h[3] = (_Float16)v.w;
  *(uint2*)(dst + i) = pk.u;
}

// ---------------------------------------------------------------------------
// Kernel 0: w1 = weight @ a1, w2 = weight @ a2   (folds Wh_n away entirely)
// ---------------------------------------------------------------------------
__global__ void gat_wvec(const float* __restrict__ weight,
                         const float* __restrict__ att,
                         float* __restrict__ w1, float* __restrict__ w2) {
  int i = blockIdx.x * blockDim.x + threadIdx.x;
  if (i >= F_DIM) return;
  const float* row = weight + (size_t)i * F_DIM;
  float acc1 = 0.f, acc2 = 0.f;
  for (int j = 0; j < F_DIM; ++j) {
    float w = row[j];
    acc1 += w * att[j];
    acc2 += w * att[F_DIM + j];
  }
  w1[i] = acc1; w2[i] = acc2;
}

// ---------------------------------------------------------------------------
// Kernel 1: s_n = node_feats @ w1 ; s_m = neighbor_feats @ w2  (one wave/row)
// ---------------------------------------------------------------------------
__global__ void gat_rowdot(const float* __restrict__ node,
                           const float* __restrict__ neigh,
                           const float* __restrict__ w1,
                           const float* __restrict__ w2,
                           float* __restrict__ s_n, float* __restrict__ s_m) {
  int gw   = blockIdx.x * (blockDim.x >> 5) + (threadIdx.x >> 5);
  int lane = threadIdx.x & 31;
  const float *src, *vec; float* dst; int row;
  if (gw < N_NODES) { src = node;  vec = w1; dst = s_n; row = gw; }
  else              { src = neigh; vec = w2; dst = s_m; row = gw - N_NODES; }
  const float* p = src + (size_t)row * F_DIM;
  float acc = 0.f;
#pragma unroll 4
  for (int f = lane; f < F_DIM; f += 32) acc += p[f] * vec[f];
#pragma unroll
  for (int off = 16; off > 0; off >>= 1) acc += __shfl_xor(acc, off, 32);
  if (lane == 0) dst[row] = acc;
}

// ---------------------------------------------------------------------------
// Kernel 2: smax = max_m s_m  (softmax upper-bound stabilizer)
// ---------------------------------------------------------------------------
__global__ void gat_smax(const float* __restrict__ s_m, float* __restrict__ smax) {
  __shared__ float red[256];
  float m = -3.4e38f;
  for (int i = threadIdx.x; i < M_NEIGH; i += 256) m = fmaxf(m, s_m[i]);
  red[threadIdx.x] = m;
  __syncthreads();
#pragma unroll
  for (int s = 128; s > 0; s >>= 1) {
    if ((int)threadIdx.x < s) red[threadIdx.x] = fmaxf(red[threadIdx.x], red[threadIdx.x + s]);
    __syncthreads();
  }
  if (threadIdx.x == 0) *smax = red[0];
}

// ---------------------------------------------------------------------------
// Kernel 3: Wh_m = neigh16 @ w16 (f16 in, f16 out), pure-WMMA inner loop.
// One 16x16 C tile per wave, K=512 in 16 steps; no in-loop conversions.
// ---------------------------------------------------------------------------
__global__ void gat_whm(const _Float16* __restrict__ neigh16,
                        const _Float16* __restrict__ w16,
                        _Float16* __restrict__ whm) {
  const int FT = F_DIM / 16;                         // 32 F-tiles
  int gw   = blockIdx.x * (blockDim.x >> 5) + (threadIdx.x >> 5);
  int lane = threadIdx.x & 31;
  int mt = gw / FT, ft = gw % FT;
  int m0 = mt * 16, f0 = ft * 16;
  int r = lane & 15, hi = lane >> 4;

  const _Float16* arow = neigh16 + (size_t)(m0 + r) * F_DIM;
  v8f acc = {};
#pragma unroll 4
  for (int kb = 0; kb < F_DIM / 32; ++kb) {
    int k0 = kb * 32;
    // A: lane<16 holds K {0..7,16..23}; lane>=16 holds {8..15,24..31}
    int base0 = k0 + hi * 8;
    v8h a0 = *(const v8h*)(arow + base0);        // 16B b128
    v8h a1 = *(const v8h*)(arow + base0 + 16);   // 16B b128
    v16h A;
#pragma unroll
    for (int e = 0; e < 8; ++e) { A[e] = a0[e]; A[8 + e] = a1[e]; }
    // B: lane holds col f0+r; K {0..7,16..23} (lane<16) / {8..15,24..31} (lane>=16)
    v16h B;
#pragma unroll
    for (int e = 0; e < 8; ++e) {
      B[e]     = w16[(size_t)(k0 + hi * 8 + e) * F_DIM + f0 + r];
      B[8 + e] = w16[(size_t)(k0 + 16 + hi * 8 + e) * F_DIM + f0 + r];
    }
    acc = __builtin_amdgcn_wmma_f32_16x16x32_f16(false, A, false, B,
                                                 (short)0, acc, false, false);
  }
  // C layout: VGPR j -> row j + 8*hi, col = lane&15
#pragma unroll
  for (int j = 0; j < 8; ++j)
    whm[(size_t)(m0 + j + hi * 8) * F_DIM + f0 + r] = (_Float16)acc[j];
}

// ---------------------------------------------------------------------------
// Kernel 4: flash-style  out = softmax(masked lrelu scores) @ Wh_m
// Block = 16 node rows, 8 waves split F (64 cols = 4 C tiles each).
// P tile produced cooperatively into LDS (2 exps/thread/iter, mask read once),
// double-buffered with ONE barrier per iteration. No online rescale:
// B_r = lrelu(s_n[r] + smax) upper-bounds every score (lrelu is monotone).
// ---------------------------------------------------------------------------
__global__ void gat_attn(const float* __restrict__ s_n,
                         const float* __restrict__ s_m,
                         const int* __restrict__ mask,
                         const _Float16* __restrict__ whm,
                         const float* __restrict__ smaxp,
                         float* __restrict__ out) {
  __shared__ _Float16 plds[2][16 * PSTRIDE];   // 2 x 1.25 KB P tiles (padded)
  __shared__ float red[256];
  __shared__ float linv_s[16];

  int t    = threadIdx.x;
  int n0   = blockIdx.x * 16;
  int wv   = t >> 5;                 // 0..7 -> F slice
  int lane = t & 31;
  int r    = lane & 15, hi = lane >> 4;
  int fbase = wv * 64;

  // -------- producer role: thread t owns row t>>4, cols 2*(t&15)..+1
  int prow  = t >> 4;
  int cpair = (t & 15) * 2;
  const int* mrow = mask + (size_t)(n0 + prow) * M_NEIGH;
  float smax = *smaxp;
  float snr  = s_n[n0 + prow];
  float lb   = snr + smax;
  float Bb   = lb > 0.f ? lb : ALPHA * lb;     // stabilizer (upper bound)
  _Float16* myp = &plds[0][prow * PSTRIDE + cpair];
  const int pofs = 16 * PSTRIDE;               // halves between buffers

  v8f acc0 = {}, acc1 = {}, acc2 = {}, acc3 = {};
  float partial = 0.f;

  // prologue: fill buffer 0
  partial += p_block(mrow, s_m, 0, cpair, snr, Bb, myp);

  const int NB = M_NEIGH / 32;                 // 256 m-blocks
#pragma unroll 2
  for (int mb = 0; mb < NB; ++mb) {
    int cur = mb & 1;
    __syncthreads();                           // P[cur] visible to all waves
    if (mb + 1 < NB)                           // produce next tile into buf^1
      partial += p_block(mrow, s_m, (mb + 1) * 32, cpair, snr, Bb,
                         myp + (cur ? 0 : pofs));
    // -------- consumer role: A from LDS (2 x ds_load_b128, bank-padded)
    const _Float16* prow_lds = &plds[cur][r * PSTRIDE + hi * 8];
    v8h a0 = *(const v8h*)(prow_lds);
    v8h a1 = *(const v8h*)(prow_lds + 16);
    v16h A;
#pragma unroll
    for (int e = 0; e < 8; ++e) { A[e] = a0[e]; A[8 + e] = a1[e]; }

    // B operand: lane holds Wh_m col fbase+16*tile+r, K in same order as A
    const _Float16* vrow = whm + (size_t)(mb * 32 + hi * 8) * F_DIM + fbase + r;
    v16h V0, V1, V2, V3;
#pragma unroll
    for (int e = 0; e < 8; ++e) {
      const _Float16* q0 = vrow + (size_t)e * F_DIM;
      const _Float16* q1 = vrow + (size_t)(16 + e) * F_DIM;
      V0[e] = q0[0];  V0[8 + e] = q1[0];
      V1[e] = q0[16]; V1[8 + e] = q1[16];
      V2[e] = q0[32]; V2[8 + e] = q1[32];
      V3[e] = q0[48]; V3[8 + e] = q1[48];
    }

    acc0 = __builtin_amdgcn_wmma_f32_16x16x32_f16(false, A, false, V0, (short)0, acc0, false, false);
    acc1 = __builtin_amdgcn_wmma_f32_16x16x32_f16(false, A, false, V1, (short)0, acc1, false, false);
    acc2 = __builtin_amdgcn_wmma_f32_16x16x32_f16(false, A, false, V2, (short)0, acc2, false, false);
    acc3 = __builtin_amdgcn_wmma_f32_16x16x32_f16(false, A, false, V3, (short)0, acc3, false, false);
  }

  // -------- row-sum reduction: partials for row R live in threads R*16..R*16+15
  __syncthreads();
  red[t] = partial;
  __syncthreads();
  if (t < 16) {
    float l = 0.f;
#pragma unroll
    for (int j = 0; j < 16; ++j) l += red[t * 16 + j];
    linv_s[t] = 1.0f / l;
  }
  __syncthreads();

#pragma unroll
  for (int j = 0; j < 8; ++j) {
    int rowj  = j + hi * 8;                    // C layout row for VGPR j
    float inv = linv_s[rowj];
    size_t o  = (size_t)(n0 + rowj) * F_DIM + fbase + r;
    out[o]      = acc0[j] * inv;
    out[o + 16] = acc1[j] * inv;
    out[o + 32] = acc2[j] * inv;
    out[o + 48] = acc3[j] * inv;
  }
}

// ---------------------------------------------------------------------------
// Launch
// ---------------------------------------------------------------------------
extern "C" void kernel_launch(void* const* d_in, const int* in_sizes, int n_in,
                              void* d_out, int out_size, void* d_ws, size_t ws_size,
                              hipStream_t stream) {
  const float* node   = (const float*)d_in[0];
  const float* neigh  = (const float*)d_in[1];
  const float* weight = (const float*)d_in[2];
  const float* attw   = (const float*)d_in[3];
  const int*   mask   = (const int*)d_in[4];
  float* out = (float*)d_out;

  char* ws = (char*)d_ws;
  _Float16* whm     = (_Float16*)ws;                            // 8 MiB
  _Float16* neigh16 = whm + (size_t)M_NEIGH * F_DIM;            // 8 MiB
  _Float16* w16     = neigh16 + (size_t)M_NEIGH * F_DIM;        // 0.5 MiB
  float* s_n  = (float*)(w16 + (size_t)F_DIM * F_DIM);
  float* s_m  = s_n + N_NODES;
  float* w1   = s_m + M_NEIGH;
  float* w2   = w1 + F_DIM;
  float* smax = w2 + F_DIM;

  const int NE_N = M_NEIGH * F_DIM;   // 4194304
  const int NE_W = F_DIM * F_DIM;     // 262144

  gat_cvt   <<<NE_N / 1024, 256, 0, stream>>>(neigh,  neigh16, NE_N);
  gat_cvt   <<<NE_W / 1024, 256, 0, stream>>>(weight, w16,     NE_W);
  gat_wvec  <<<2, 256, 0, stream>>>(weight, attw, w1, w2);
  gat_rowdot<<<(N_NODES + M_NEIGH) / 8, 256, 0, stream>>>(node, neigh, w1, w2, s_n, s_m);
  gat_smax  <<<1, 256, 0, stream>>>(s_m, smax);
  gat_whm   <<<(M_NEIGH / 16) * (F_DIM / 16) / 8, 256, 0, stream>>>(neigh16, w16, whm);
  gat_attn  <<<N_NODES / 16, 256, 0, stream>>>(s_n, s_m, mask, whm, smax, out);
}
